// StandardAttention_82291573391983
// MI455X (gfx1250) — matrix operand order repaired and checked by reading, not compiled, not run on hardware
//
#include <hip/hip_runtime.h>
#include <hip/hip_bf16.h>
#include <math.h>
#include <stdint.h>

// ---------------------------------------------------------------------------
// StandardAttention for MI455X (gfx1250).
// Pipeline: [x->bf16, W->bf16^T] prep; QKV GEMM; RoPE (Q pre-scaled by
// 1/sqrt(d)); causal GQA flash attention; output GEMM.
// Matmuls: v_wmma_f32_16x16x32_bf16 (wave32). LDS staging:
// global_load_async_to_lds_b128 (ASYNCcnt), double-buffered. Softmax
// reductions: DPP butterflies (no LDS).
// ---------------------------------------------------------------------------

typedef __attribute__((ext_vector_type(16))) __bf16 v16bf;
typedef __attribute__((ext_vector_type(8)))  float  v8f;

union FragU {
  uint4 u4[2];
  v16bf bf;
};

#define SEQ   2048
#define DIMM  4096
#define QKVN  6144   // 32*128 q + 8*128 k + 8*128 v
#define HD    128
#define NH    32
#define NKV   8

__device__ __forceinline__ v8f wmma_bf16(v16bf a, v16bf b, v8f c) {
  return __builtin_amdgcn_wmma_f32_16x16x32_bf16(false, a, false, b, (short)0, c,
                                                 false, false);
}

__device__ __forceinline__ unsigned lds_off(const void* p) {
  // LDS aperture in flat addressing: LDS offset is the low 32 bits of the address
  return (unsigned)(uintptr_t)p;
}

__device__ __forceinline__ void async_b128(unsigned lds, const void* gp) {
  asm volatile("global_load_async_to_lds_b128 %0, %1, off"
               :: "v"(lds), "v"((unsigned long long)(uintptr_t)gp) : "memory");
}

// 16-lane (half-wave) reductions via DPP: quad xor1, xor2, then row_ror 4/8.
template <int CTRL>
__device__ __forceinline__ float dpp_f(float v) {
  return __int_as_float(__builtin_amdgcn_update_dpp(
      0, __float_as_int(v), CTRL, 0xF, 0xF, true));
}
__device__ __forceinline__ float red16_max(float v) {
  v = fmaxf(v, dpp_f<0x0B1>(v));   // quad_perm(1,0,3,2)
  v = fmaxf(v, dpp_f<0x04E>(v));   // quad_perm(2,3,0,1)
  v = fmaxf(v, dpp_f<0x124>(v));   // row_ror:4
  v = fmaxf(v, dpp_f<0x128>(v));   // row_ror:8
  return v;
}
__device__ __forceinline__ float red16_sum(float v) {
  v += dpp_f<0x0B1>(v);
  v += dpp_f<0x04E>(v);
  v += dpp_f<0x124>(v);
  v += dpp_f<0x128>(v);
  return v;
}

// ---------------------------------------------------------------------------
// Prep: f32 -> bf16 (row-major copy), 4 elements/thread.
// ---------------------------------------------------------------------------
__global__ void f32_to_bf16(const float* __restrict__ src,
                            __bf16* __restrict__ dst, int n) {
  int i = (blockIdx.x * blockDim.x + threadIdx.x) * 4;
  if (i >= n) return;
  float4 f = *(const float4*)(src + i);
  __bf16 t[4] __attribute__((aligned(8)));
  t[0] = (__bf16)f.x; t[1] = (__bf16)f.y; t[2] = (__bf16)f.z; t[3] = (__bf16)f.w;
  *(uint2*)(dst + i) = *(const uint2*)t;
}

// ---------------------------------------------------------------------------
// Prep: W[K][N] f32 -> Wt[N][K] bf16, tiled transpose (32x32, 256 threads).
// ---------------------------------------------------------------------------
__global__ void __launch_bounds__(256)
transpose_to_bf16(const float* __restrict__ W, __bf16* __restrict__ Wt,
                  int K, int N) {
  __shared__ float tile[32][33];
  const int c = threadIdx.x;        // 0..31
  const int r = threadIdx.y;        // 0..7
  const int n0 = blockIdx.x * 32;
  const int k0 = blockIdx.y * 32;
#pragma unroll
  for (int i = 0; i < 4; ++i)
    tile[r + 8 * i][c] = W[(size_t)(k0 + r + 8 * i) * N + (n0 + c)];
  __syncthreads();
#pragma unroll
  for (int i = 0; i < 4; ++i)
    Wt[(size_t)(n0 + r + 8 * i) * K + (k0 + c)] = (__bf16)tile[c][r + 8 * i];
}

// ---------------------------------------------------------------------------
// C = A(MxK,bf16 row-major) * Bt(NxK,bf16)^T -> C(MxN,f32). Block 64x128,
// 4 waves; B tile async-staged (double-buffered); A frags direct from global.
// ---------------------------------------------------------------------------
__device__ __forceinline__ void stage_b_tile(__bf16 (*Bs)[56],
                                             const __bf16* src, int K,
                                             int tid) {
#pragma unroll
  for (int it = 0; it < 4; ++it) {
    int idx = it * 128 + tid;        // 512 x 16B segments = 128n x 32k
    int row = idx >> 2, seg = idx & 3;
    async_b128(lds_off(&Bs[row][seg * 8]), src + (size_t)row * K + seg * 8);
  }
}

template <int N, int K>
__global__ void __launch_bounds__(128)
gemm_bf16t(const __bf16* __restrict__ A, const __bf16* __restrict__ Bt,
           float* __restrict__ C) {
  __shared__ __bf16 Bs[2][128][56];   // [n][k], pitch 112B: aligned, no conflicts

  const int tid  = threadIdx.x;
  const int wave = tid >> 5;
  const int lane = tid & 31;
  const int half = lane >> 4;
  const int l16  = lane & 15;
  const int m0   = blockIdx.y * 64;
  const int n0   = blockIdx.x * 128;
  const int mrow = m0 + wave * 16 + l16;

  const __bf16* bsrc = Bt + (size_t)n0 * K;
  const __bf16* arow = A + (size_t)mrow * K;

  v8f acc[8] = {};
  constexpr int nk = K / 32;

  stage_b_tile(Bs[0], bsrc, K, tid);

  for (int s = 0; s < nk; ++s) {
    asm volatile("s_wait_asynccnt 0" ::: "memory");
    __syncthreads();
    if (s + 1 < nk) stage_b_tile(Bs[(s + 1) & 1], bsrc + (s + 1) * 32, K, tid);

    // A fragment (16x32): e -> K = e + 8*((e>>3)+half): two 8-bf16 runs
    FragU af;
    af.u4[0] = *(const uint4*)(arow + s * 32 + 8 * half);
    af.u4[1] = *(const uint4*)(arow + s * 32 + 16 + 8 * half);
    if (s + 2 < nk) __builtin_prefetch(arow + (s + 2) * 32, 0, 1);

    __bf16 (*B)[56] = Bs[s & 1];
#pragma unroll
    for (int j = 0; j < 8; ++j) {
      FragU bfg;                      // B frag (32x16): N=lane%16, K=e+8*((e>>3)+half)
      const __bf16* bp = &B[j * 16 + l16][8 * half];
      bfg.u4[0] = *(const uint4*)bp;
      bfg.u4[1] = *(const uint4*)(bp + 16);
      acc[j] = wmma_bf16(af.bf, bfg.bf, acc[j]);
    }
  }

  // C/D layout: m = r + 8*half, n = lane%16; hoisted row pointers
#pragma unroll
  for (int r = 0; r < 8; ++r) {
    float* cp = C + (size_t)(m0 + wave * 16 + r + 8 * half) * N + n0 + l16;
#pragma unroll
    for (int j = 0; j < 8; ++j) cp[j * 16] = acc[j][r];
  }
}

// ---------------------------------------------------------------------------
// RoPE + head split, bf16 out: q[32][seq][128] (rope, pre-scaled 1/sqrt(d)),
// k[8][seq][128] (rope), vt[8][128][seq] (transposed copy).
// ---------------------------------------------------------------------------
__global__ void rope_split(const float* __restrict__ qkv,
                           __bf16* __restrict__ q, __bf16* __restrict__ k,
                           __bf16* __restrict__ vt) {
  int idx = blockIdx.x * blockDim.x + threadIdx.x;
  const int total = 48 * SEQ * 64;
  if (idx >= total) return;
  const int i = idx & 63;
  const int s = (idx >> 6) & (SEQ - 1);
  const int h = idx / (SEQ * 64);
  const float qscale = 0.08838834764831845f;   // 1/sqrt(128)

  if (h < NH + NKV) {                // rope path (q or k)
    const float* src;
    __bf16* dst;
    float sc;
    if (h < NH) {
      src = qkv + (size_t)s * QKVN + h * HD;
      dst = q + ((size_t)h * SEQ + s) * HD;
      sc = qscale;                   // fold score scale into Q
    } else {
      int kh = h - NH;
      src = qkv + (size_t)s * QKVN + NH * HD + kh * HD;
      dst = k + ((size_t)kh * SEQ + s) * HD;
      sc = 1.0f;
    }
    float x1 = src[2 * i], x2 = src[2 * i + 1];
    float inv = __powf(10000.0f, -(2.0f * (float)i) / 128.0f);
    float ang = (float)s * inv;
    float sn, cs;
    __sincosf(ang, &sn, &cs);
    dst[2 * i]     = (__bf16)(sc * (x1 * cs - x2 * sn));
    dst[2 * i + 1] = (__bf16)(sc * (x1 * sn + x2 * cs));
  } else {                           // v: transposed copy vt[kvh][d][s]
    int vh = h - NH - NKV;
    const float* src = qkv + (size_t)s * QKVN + (NH + NKV) * HD + vh * HD;
    vt[((size_t)vh * HD + 2 * i) * SEQ + s]     = (__bf16)src[2 * i];
    vt[((size_t)vh * HD + 2 * i + 1) * SEQ + s] = (__bf16)src[2 * i + 1];
  }
}

// ---------------------------------------------------------------------------
// Flash attention: grid (head, seq/64), 4 waves x 16 query rows. K/V async-
// staged per 32-key block (double-buffered); online softmax in WMMA C-layout
// with DPP reductions; bf16 output feeds the final GEMM.
// ---------------------------------------------------------------------------
__device__ __forceinline__ void stage_kv(__bf16 (*Ks)[136], __bf16 (*Vs)[56],
                                         const __bf16* kbase,
                                         const __bf16* vbase, int tid) {
#pragma unroll
  for (int it = 0; it < 4; ++it) {   // K: 32 key rows x 128 d (16 segs/row)
    int idx = it * 128 + tid;
    int row = idx >> 4, seg = idx & 15;
    async_b128(lds_off(&Ks[row][seg * 8]), kbase + (size_t)row * HD + seg * 8);
  }
#pragma unroll
  for (int it = 0; it < 4; ++it) {   // V^T: 128 d rows x 32 keys (4 segs/row)
    int idx = it * 128 + tid;
    int row = idx >> 2, seg = idx & 3;
    async_b128(lds_off(&Vs[row][seg * 8]), vbase + (size_t)row * SEQ + seg * 8);
  }
}

__global__ void __launch_bounds__(128)
flash_attn(const __bf16* __restrict__ q, const __bf16* __restrict__ k,
           const __bf16* __restrict__ vt, __bf16* __restrict__ out) {
  __shared__ __bf16 Ks[2][32][136];   // [key][d], pitch 272B
  __shared__ __bf16 Vst[2][128][56];  // [d][key], pitch 112B
  __shared__ __bf16 Ps[4][16][56];    // per-wave P bounce tile [m][key]

  const int head = blockIdx.x;
  const int qblk = blockIdx.y;
  const int kvh  = head >> 2;
  const int tid  = threadIdx.x;
  const int wave = tid >> 5;
  const int lane = tid & 31;
  const int half = lane >> 4;
  const int l16  = lane & 15;

  const int qbase = qblk * 64;
  const int qrow  = qbase + wave * 16 + l16;

  const __bf16* kh_base = k + (size_t)kvh * SEQ * HD;
  const __bf16* vh_base = vt + (size_t)kvh * HD * SEQ;

  // Q fragments (already scaled by 1/sqrt(d)).
  v16bf qf[4];
  {
    const __bf16* qp = q + ((size_t)head * SEQ + qrow) * HD;
#pragma unroll
    for (int c = 0; c < 4; ++c) {
      FragU f;
      f.u4[0] = *(const uint4*)(qp + c * 32 + 8 * half);
      f.u4[1] = *(const uint4*)(qp + c * 32 + 16 + 8 * half);
      qf[c] = f.bf;
    }
  }

  v8f acc_o[8] = {};
  float m_st[8], l_st[8];
#pragma unroll
  for (int r = 0; r < 8; ++r) { m_st[r] = -1e30f; l_st[r] = 0.0f; }

  const int nkb = (qbase + 63) / 32 + 1;       // uniform over workgroup

  stage_kv(Ks[0], Vst[0], kh_base, vh_base, tid);

  for (int kb = 0; kb < nkb; ++kb) {
    asm volatile("s_wait_asynccnt 0" ::: "memory");
    __syncthreads();
    if (kb + 1 < nkb)
      stage_kv(Ks[(kb + 1) & 1], Vst[(kb + 1) & 1],
               kh_base + (size_t)(kb + 1) * 32 * HD,
               vh_base + (kb + 1) * 32, tid);

    __bf16 (*Kb)[136] = Ks[kb & 1];
    __bf16 (*Vb)[56]  = Vst[kb & 1];

    // S = Q (16x128) x K^T (128x32): two 16-key tiles, 4 d-chunks.
    v8f s0 = {}, s1 = {};
#pragma unroll
    for (int c = 0; c < 4; ++c) {
      FragU b0, b1;
      const __bf16* kp0 = &Kb[l16][c * 32 + 8 * half];
      const __bf16* kp1 = &Kb[16 + l16][c * 32 + 8 * half];
      b0.u4[0] = *(const uint4*)kp0; b0.u4[1] = *(const uint4*)(kp0 + 16);
      b1.u4[0] = *(const uint4*)kp1; b1.u4[1] = *(const uint4*)(kp1 + 16);
      s0 = wmma_bf16(qf[c], b0.bf, s0);
      s1 = wmma_bf16(qf[c], b1.bf, s1);
    }

    // Online softmax per row m = r + 8*half (replicated across 16 lanes);
    // 16-lane reductions fully in VALU via DPP.
    const int key0 = kb * 32 + l16;
    const int key1 = key0 + 16;
    float p0v[8], p1v[8];
#pragma unroll
    for (int r = 0; r < 8; ++r) {
      int row = qbase + wave * 16 + r + 8 * half;
      float a0 = (key0 > row) ? -1e30f : s0[r];
      float a1 = (key1 > row) ? -1e30f : s1[r];
      float mx = red16_max(fmaxf(a0, a1));
      float mnew = fmaxf(m_st[r], mx);
      float al   = __expf(m_st[r] - mnew);
      float p0   = __expf(a0 - mnew);
      float p1   = __expf(a1 - mnew);
      float rs   = red16_sum(p0 + p1);
      l_st[r] = l_st[r] * al + rs;
      m_st[r] = mnew;
#pragma unroll
      for (int t = 0; t < 8; ++t) acc_o[t][r] *= al;
      p0v[r] = p0;
      p1v[r] = p1;
    }

    // P: C-layout -> A-fragment layout through per-wave LDS tile.
#pragma unroll
    for (int r = 0; r < 8; ++r) {
      Ps[wave][r + 8 * half][l16]      = (__bf16)p0v[r];
      Ps[wave][r + 8 * half][16 + l16] = (__bf16)p1v[r];
    }
    asm volatile("s_wait_dscnt 0" ::: "memory");
    FragU pf;
    {
      const __bf16* pp = &Ps[wave][l16][0];
      pf.u4[0] = *(const uint4*)(pp + 8 * half);
      pf.u4[1] = *(const uint4*)(pp + 16 + 8 * half);
    }

    // O += P (16x32) x V (32x128): 8 d-tiles, contiguous B-frags from Vst.
#pragma unroll
    for (int t = 0; t < 8; ++t) {
      FragU bv;
      const __bf16* vp = &Vb[t * 16 + l16][8 * half];
      bv.u4[0] = *(const uint4*)vp;
      bv.u4[1] = *(const uint4*)(vp + 16);
      acc_o[t] = wmma_bf16(pf.bf, bv.bf, acc_o[t]);
    }
  }

  // Epilogue: out[row][head*128 + d] = O / l (bf16), hoisted row pointers.
#pragma unroll
  for (int r = 0; r < 8; ++r) {
    float rl = 1.0f / l_st[r];
    __bf16* op = out + (size_t)(qbase + wave * 16 + r + 8 * half) * (NH * HD) +
                 head * HD + l16;
#pragma unroll
    for (int t = 0; t < 8; ++t) op[t * 16] = (__bf16)(acc_o[t][r] * rl);
  }
}

// ---------------------------------------------------------------------------
extern "C" void kernel_launch(void* const* d_in, const int* in_sizes, int n_in,
                              void* d_out, int out_size, void* d_ws,
                              size_t ws_size, hipStream_t stream) {
  const float* x     = (const float*)d_in[0];   // [1, 2048, 4096]
  const float* w_qkv = (const float*)d_in[1];   // [4096, 6144]
  const float* w_o   = (const float*)d_in[2];   // [4096, 4096]
  float* out = (float*)d_out;                   // [1, 2048, 4096]

  __bf16* xb    = (__bf16*)d_ws;                        // 2048*4096
  __bf16* wqkvt = xb + (size_t)SEQ * DIMM;              // 6144*4096
  __bf16* wot   = wqkvt + (size_t)QKVN * DIMM;          // 4096*4096
  __bf16* qr    = wot + (size_t)DIMM * DIMM;            // 32*2048*128
  __bf16* kr    = qr + (size_t)NH * SEQ * HD;           // 8*2048*128
  __bf16* vtr   = kr + (size_t)NKV * SEQ * HD;          // 8*128*2048
  __bf16* attnb = vtr + (size_t)NKV * SEQ * HD;         // 2048*4096
  float*  qkv   = (float*)(attnb + (size_t)SEQ * DIMM); // 2048*6144 f32

  // Prep: bf16 conversion + weight transposes.
  {
    int n = SEQ * DIMM;
    f32_to_bf16<<<(n / 4 + 255) / 256, 256, 0, stream>>>(x, xb, n);
  }
  transpose_to_bf16<<<dim3(QKVN / 32, DIMM / 32), dim3(32, 8), 0, stream>>>(
      w_qkv, wqkvt, DIMM, QKVN);
  transpose_to_bf16<<<dim3(DIMM / 32, DIMM / 32), dim3(32, 8), 0, stream>>>(
      w_o, wot, DIMM, DIMM);

  dim3 blk(128);

  // 1) qkv = x @ w_qkv   (2048 x 4096 x 6144)
  gemm_bf16t<QKVN, DIMM><<<dim3(QKVN / 128, SEQ / 64), blk, 0, stream>>>(
      xb, wqkvt, qkv);

  // 2) rope + split heads (bf16 out, Q pre-scaled, V transposed)
  {
    int total = 48 * SEQ * 64;
    rope_split<<<(total + 255) / 256, 256, 0, stream>>>(qkv, qr, kr, vtr);
  }

  // 3) causal GQA flash attention (bf16 out)
  flash_attn<<<dim3(NH, SEQ / 64), blk, 0, stream>>>(qr, kr, vtr, attnb);

  // 4) out = attn @ w_o  (2048 x 4096 x 4096)
  gemm_bf16t<DIMM, DIMM><<<dim3(DIMM / 128, SEQ / 64), blk, 0, stream>>>(
      attnb, wot, out);
}
